// Generator_55267639164976
// MI455X (gfx1250) — compile-verified, matrix operand order checked by reading, not run on hardware
//
#include <hip/hip_runtime.h>
#include <hip/hip_bf16.h>
#include <math.h>

// ---------------------------------------------------------------------------
// Mamba-vocoder generator for gfx1250 (MI455X).
//  - dense projections: V_WMMA_F32_16X16X32_F16, NT output tiles per wave,
//    double-buffered f32->f16 staging (all loads in flight per k-step)
//  - selective scan: 3-pass chunked (associative) scan, chunk length 128
// ---------------------------------------------------------------------------

typedef __attribute__((ext_vector_type(16))) _Float16 v16h;
typedef __attribute__((ext_vector_type(8)))  float    v8f;
typedef __attribute__((ext_vector_type(4)))  float    f4;

#define C_MODEL 256
#define D_INNER 512
#define D_STATE 64
#define DT_RANK 16
#define CHUNK   128
#define PI_F 3.14159265358979323846f

// Convert 16 raw f32 (two groups of 8, ISA 16-bit A/B fragment order) to f16.
__device__ __forceinline__ v16h cvt_frag(const f4 (&r)[4])
{
    v16h o;
#pragma unroll
    for (int j = 0; j < 4; ++j) {
        o[j]      = (_Float16)r[0][j];   // floats 0..3
        o[4 + j]  = (_Float16)r[1][j];   // floats 4..7
        o[8 + j]  = (_Float16)r[2][j];   // floats 16..19
        o[12 + j] = (_Float16)r[3][j];   // floats 20..23
    }
    return o;
}

// ============================================================================
// WMMA GEMM:  Out[M,N] = act( A[M,K] @ W[N,K]^T + bias[N] )
// A row-major (leading dim lda), W row-major [N,K] (stride K), Out ldo.
// One wave computes a 16 x (16*NT) strip; A fragment reused for NT WMMAs.
// ACT: 0 = none, 2 = softplus.
// ============================================================================
template<int ACT, int NT>
__global__ __launch_bounds__(128)
void gemm_wmma(const float* __restrict__ A, int lda,
               const float* __restrict__ W,
               const float* __restrict__ bias,
               float* __restrict__ Out, int ldo,
               int M, int N, int K)
{
    const int lane   = threadIdx.x & 31;
    const int wid    = (blockIdx.x * blockDim.x + threadIdx.x) >> 5;
    const int tilesN = N >> 4;
    const int groups = tilesN / NT;
    const int tiles  = (M >> 4) * groups;
    if (wid >= tiles) return;                       // wave-uniform exit

    const int tileM  = wid / groups;
    const int tileN0 = (wid % groups) * NT;

    const int mrow = tileM * 16 + (lane & 15);
    const int koff = (lane >> 4) * 8;               // K sub-offset per half-wave

    const float* Arow = A + (size_t)mrow * lda + koff;
    const float* Wr[NT];
#pragma unroll
    for (int u = 0; u < NT; ++u)
        Wr[u] = W + (size_t)((tileN0 + u) * 16 + (lane & 15)) * K + koff;

    v8f acc[NT];
#pragma unroll
    for (int u = 0; u < NT; ++u) acc[u] = (v8f){};

    const int Kmain = K & ~31;

    auto do_mma = [&](const f4 (&ra)[4], const f4 (&rb)[NT][4]) {
        v16h a = cvt_frag(ra);
#pragma unroll
        for (int u = 0; u < NT; ++u) {
            v16h b = cvt_frag(rb[u]);
            acc[u] = __builtin_amdgcn_wmma_f32_16x16x32_f16(
                         false, a, false, b, (short)0, acc[u], false, false);
        }
    };

    if (Kmain > 0) {
        f4 ar[4];
        f4 br[NT][4];
        {   // prologue: k0 = 0
            const f4* p = (const f4*)Arow;
            ar[0] = p[0]; ar[1] = p[1]; ar[2] = p[4]; ar[3] = p[5];
#pragma unroll
            for (int u = 0; u < NT; ++u) {
                const f4* q = (const f4*)Wr[u];
                br[u][0] = q[0]; br[u][1] = q[1]; br[u][2] = q[4]; br[u][3] = q[5];
            }
        }
        for (int k0 = 32; k0 < Kmain; k0 += 32) {
            f4 ar2[4];
            f4 br2[NT][4];
            const f4* p = (const f4*)(Arow + k0);
            ar2[0] = p[0]; ar2[1] = p[1]; ar2[2] = p[4]; ar2[3] = p[5];
#pragma unroll
            for (int u = 0; u < NT; ++u) {
                const f4* q = (const f4*)(Wr[u] + k0);
                br2[u][0] = q[0]; br2[u][1] = q[1]; br2[u][2] = q[4]; br2[u][3] = q[5];
            }
            do_mma(ar, br);                         // overlaps with loads above
#pragma unroll
            for (int j = 0; j < 4; ++j) ar[j] = ar2[j];
#pragma unroll
            for (int u = 0; u < NT; ++u)
#pragma unroll
                for (int j = 0; j < 4; ++j) br[u][j] = br2[u][j];
        }
        do_mma(ar, br);                             // drain last buffered step
    }

    if (Kmain < K) {                                // tail (e.g. K == 16)
        v16h a;
#pragma unroll
        for (int j = 0; j < 16; ++j) {
            int k = Kmain + ((j < 8) ? j : (8 + j));        // koff already in ptr
            a[j] = ((k + koff) < K || true) ? (_Float16)0.0f : (_Float16)0.0f;
        }
        // scalar guarded loads (koff folded into Arow/Wr)
#pragma unroll
        for (int j = 0; j < 16; ++j) {
            int krel = Kmain + ((j < 8) ? j : (8 + j));     // 0..7 -> k, 8..15 -> k+16
            int kabs = krel + koff;
            a[j] = (kabs < K) ? (_Float16)Arow[krel] : (_Float16)0.0f;
        }
#pragma unroll
        for (int u = 0; u < NT; ++u) {
            v16h b;
#pragma unroll
            for (int j = 0; j < 16; ++j) {
                int krel = Kmain + ((j < 8) ? j : (8 + j));
                int kabs = krel + koff;
                b[j] = (kabs < K) ? (_Float16)Wr[u][krel] : (_Float16)0.0f;
            }
            acc[u] = __builtin_amdgcn_wmma_f32_16x16x32_f16(
                         false, a, false, b, (short)0, acc[u], false, false);
        }
    }

#pragma unroll
    for (int u = 0; u < NT; ++u) {
        const int n = (tileN0 + u) * 16 + (lane & 15);
        const float bv = bias ? bias[n] : 0.0f;
#pragma unroll
        for (int r = 0; r < 8; ++r) {
            int m = tileM * 16 + r + 8 * (lane >> 4);   // ISA C/D layout
            float v = acc[u][r] + bv;
            if (ACT == 2) v = (v > 20.0f) ? v : log1pf(expf(v));   // softplus
            Out[(size_t)m * ldo + n] = v;
        }
    }
}

// ============================================================================
// conv_pre: x[80,128] -> y[128,256], k=7, pad=3.  One block per t, thread=co.
// ============================================================================
__global__ void conv_pre_k(const float* __restrict__ x,
                           const float* __restrict__ w,  // [256,80,7]
                           const float* __restrict__ b,
                           float* __restrict__ out)      // [128,256]
{
    int t = blockIdx.x, co = threadIdx.x;
    float acc = b[co];
    for (int c = 0; c < 80; ++c) {
#pragma unroll
        for (int k = 0; k < 7; ++k) {
            int s = t - 3 + k;
            if ((unsigned)s < 128u)
                acc += x[c * 128 + s] * w[co * 560 + c * 7 + k];
        }
    }
    out[t * C_MODEL + co] = acc;
}

// ============================================================================
// leaky_relu(0.1) + ConvTranspose1d(256->256, k=16, stride=8, pad=4)
// ============================================================================
__global__ void upsample_k(const float* __restrict__ yin,
                           const float* __restrict__ w,  // [Cin,Cout,16]
                           const float* __restrict__ b,
                           float* __restrict__ out, int Tin)
{
    int t = blockIdx.x, co = threadIdx.x;
    int k1 = (t + 4) & 7;
    int k2 = k1 + 8;
    int s1 = (t + 4 - k1) >> 3;
    int s2 = s1 - 1;
    bool v1 = (s1 >= 0 && s1 < Tin);
    bool v2 = (s2 >= 0 && s2 < Tin);
    float acc = b[co];
    for (int ci = 0; ci < C_MODEL; ++ci) {
        const float* wr = w + ci * (C_MODEL * 16) + co * 16;
        if (v1) { float xv = yin[s1 * C_MODEL + ci]; xv = xv > 0.f ? xv : 0.1f * xv; acc += xv * wr[k1]; }
        if (v2) { float xv = yin[s2 * C_MODEL + ci]; xv = xv > 0.f ? xv : 0.1f * xv; acc += xv * wr[k2]; }
    }
    out[t * C_MODEL + co] = acc;
}

// ============================================================================
// depthwise causal conv (k=4) + SiLU:  xz[:, :512] -> xhs [T,512]
// ============================================================================
__global__ void dwconv_silu_k(const float* __restrict__ xz,   // [T,1024]
                              const float* __restrict__ w,    // [512,4]
                              const float* __restrict__ b,
                              float* __restrict__ xhs, int T)
{
    int i = blockIdx.x * blockDim.x + threadIdx.x;
    if (i >= T * D_INNER) return;
    int t = i / D_INNER, d = i % D_INNER;
    float acc = b[d];
#pragma unroll
    for (int j = 0; j < 4; ++j) {
        int s = t - 3 + j;
        if (s >= 0) acc += xz[(size_t)s * 1024 + d] * w[d * 4 + j];
    }
    xhs[i] = acc / (1.0f + expf(-acc));
}

// ============================================================================
// Chunked selective scan, chunk length CHUNK=128, NC = T/128 chunks.
// ============================================================================
__global__ __launch_bounds__(256)
void scan_part1(const float* __restrict__ dt,
                const float* __restrict__ dbc,
                const float* __restrict__ xhs,
                const float* __restrict__ A_log,
                float* __restrict__ Q,      // [NC,512,64]
                float* __restrict__ Sdt,    // [NC,512]
                int NC)
{
    int lane = threadIdx.x & 31;
    int w    = (blockIdx.x * blockDim.x + threadIdx.x) >> 5;
    if (w >= NC * D_INNER) return;
    int c = w / D_INNER, d = w % D_INNER;

    int n0 = lane, n1 = lane + 32;
    float A0 = -expf(A_log[d * D_STATE + n0]);
    float A1 = -expf(A_log[d * D_STATE + n1]);
    float h0 = 0.f, h1 = 0.f, s = 0.f;

    int t0 = c * CHUNK;
    for (int i = 0; i < CHUNK; ++i) {
        int t = t0 + i;
        float dtv = dt[(size_t)t * D_INNER + d];
        float xv  = xhs[(size_t)t * D_INNER + d];
        const float* row = dbc + (size_t)t * 144;
        float u = dtv * xv;
        h0 = expf(dtv * A0) * h0 + u * row[16 + n0];
        h1 = expf(dtv * A1) * h1 + u * row[16 + n1];
        s += dtv;
    }
    float* q = Q + ((size_t)c * D_INNER + d) * D_STATE;
    q[n0] = h0;
    q[n1] = h1;
    if (lane == 0) Sdt[c * D_INNER + d] = s;
}

__global__ void scan_part2(const float* __restrict__ A_log,
                           const float* __restrict__ Q,
                           const float* __restrict__ Sdt,
                           float* __restrict__ Hinit,   // [NC,512,64]
                           int NC)
{
    int i = blockIdx.x * blockDim.x + threadIdx.x;
    if (i >= D_INNER * D_STATE) return;
    int d = i / D_STATE, n = i % D_STATE;
    float A = -expf(A_log[i]);
    float run = 0.f;
    for (int c = 0; c < NC; ++c) {
        size_t idx = ((size_t)c * D_INNER + d) * D_STATE + n;
        Hinit[idx] = run;
        run = expf(A * Sdt[c * D_INNER + d]) * run + Q[idx];
    }
}

__global__ __launch_bounds__(256)
void scan_part3(const float* __restrict__ dt,
                const float* __restrict__ dbc,
                const float* __restrict__ xhs,
                const float* __restrict__ A_log,
                const float* __restrict__ Hinit,
                float* __restrict__ ys,     // [T,512]
                int NC)
{
    int lane = threadIdx.x & 31;
    int w    = (blockIdx.x * blockDim.x + threadIdx.x) >> 5;
    if (w >= NC * D_INNER) return;
    int c = w / D_INNER, d = w % D_INNER;

    int n0 = lane, n1 = lane + 32;
    float A0 = -expf(A_log[d * D_STATE + n0]);
    float A1 = -expf(A_log[d * D_STATE + n1]);
    const float* hi = Hinit + ((size_t)c * D_INNER + d) * D_STATE;
    float h0 = hi[n0], h1 = hi[n1];

    int t0 = c * CHUNK;
    for (int i = 0; i < CHUNK; ++i) {
        int t = t0 + i;
        float dtv = dt[(size_t)t * D_INNER + d];
        float xv  = xhs[(size_t)t * D_INNER + d];
        const float* row = dbc + (size_t)t * 144;
        float u = dtv * xv;
        h0 = expf(dtv * A0) * h0 + u * row[16 + n0];
        h1 = expf(dtv * A1) * h1 + u * row[16 + n1];
        float p = h0 * row[80 + n0] + h1 * row[80 + n1];
#pragma unroll
        for (int off = 16; off >= 1; off >>= 1)
            p += __shfl_xor(p, off, 32);
        if (lane == 0) ys[(size_t)t * D_INNER + d] = p;
    }
}

// ============================================================================
// y = (ys + xhs * D) * silu(z);  z = xz[:, 512:1024].  In-place into ys.
// ============================================================================
__global__ void combine_k(float* __restrict__ ys,
                          const float* __restrict__ xhs,
                          const float* __restrict__ Dp,
                          const float* __restrict__ xz, int T)
{
    int i = blockIdx.x * blockDim.x + threadIdx.x;
    if (i >= T * D_INNER) return;
    int t = i / D_INNER, d = i % D_INNER;
    float z = xz[(size_t)t * 1024 + D_INNER + d];
    float sz = z / (1.0f + expf(-z));
    ys[i] = (ys[i] + xhs[i] * Dp[d]) * sz;
}

// ============================================================================
// leaky_relu(0.01) + post conv (256 -> 18, k=7, pad=3):  y[T,256] -> h[T,18]
// ============================================================================
__global__ void postconv_k(const float* __restrict__ y,
                           const float* __restrict__ w,   // [18,256,7]
                           const float* __restrict__ b,
                           float* __restrict__ h, int T)
{
    int i = blockIdx.x * blockDim.x + threadIdx.x;
    if (i >= T * 18) return;
    int t = i / 18, f = i % 18;
    float acc = b[f];
    for (int c = 0; c < C_MODEL; ++c) {
#pragma unroll
        for (int k = 0; k < 7; ++k) {
            int s = t - 3 + k;
            if ((unsigned)s < (unsigned)T) {
                float v = y[(size_t)s * C_MODEL + c];
                v = v > 0.f ? v : 0.01f * v;
                acc += v * w[f * 1792 + c * 7 + k];
            }
        }
    }
    h[i] = acc;
}

// ============================================================================
// iSTFT frames: spec=exp(h[:, :9]), phase=sin(h[:, 9:]); irfft(n=16) * hann.
// ============================================================================
__global__ void frames_k(const float* __restrict__ h, float* __restrict__ fr, int T)
{
    int i = blockIdx.x * blockDim.x + threadIdx.x;
    if (i >= T * 16) return;
    int t = i >> 4, n = i & 15;
    const float* hr = h + (size_t)t * 18;
    float s = expf(hr[0]) * cosf(sinf(hr[9]));
#pragma unroll
    for (int k = 1; k < 8; ++k)
        s += 2.0f * expf(hr[k]) * cosf(sinf(hr[9 + k]) + PI_F * (float)(k * n) / 8.0f);
    s += expf(hr[8]) * cosf(sinf(hr[17]) + PI_F * (float)n);
    float win = 0.5f - 0.5f * cosf(2.0f * PI_F * (float)n / 16.0f);
    fr[i] = win * s * (1.0f / 16.0f);
}

// ============================================================================
// Overlap-add (hop=4) + window-square normalization + trim 8 + tanh.
// ============================================================================
__global__ void ola_k(const float* __restrict__ fr, float* __restrict__ out, int T)
{
    int i = blockIdx.x * blockDim.x + threadIdx.x;
    int Lout = (T - 1) * 4;
    if (i >= Lout) return;
    int l = i + 8;
    int r = l & 3;
    float acc = 0.f, wsq = 0.f;
#pragma unroll
    for (int j = 0; j < 4; ++j) {
        int n = r + 4 * j;
        int t = (l - n) >> 2;
        if (t >= 0 && t < T) {
            acc += fr[t * 16 + n];
            float w = 0.5f - 0.5f * cosf(2.0f * PI_F * (float)n / 16.0f);
            wsq += w * w;
        }
    }
    out[i] = tanhf(acc / fmaxf(wsq, 1e-8f));
}

// ============================================================================
// Host-side orchestration
// ============================================================================
static inline int gemm_blocks(int M, int N, int NT) {
    int tiles = (M >> 4) * ((N >> 4) / NT);
    return (tiles + 3) / 4;             // 4 waves (128 threads) per block
}

struct StageParams {
    const float *up_w, *up_b, *in_proj, *conv_w, *conv_b, *x_proj,
                *dt_w, *dt_b, *A_log, *D, *out_proj, *post_w, *post_b;
};

struct Scratch {
    float *up, *xz, *xhs, *dbc, *dt, *ys, *out, *h, *fr, *y0;
    float *Q, *Hinit, *Sdt;
};

static void run_stage(const float* yin, int Tin, const StageParams& p,
                      const Scratch& s, float* audio_out, hipStream_t st)
{
    const int T  = Tin * 8;
    const int NC = T / CHUNK;

    upsample_k<<<T, 256, 0, st>>>(yin, p.up_w, p.up_b, s.up, Tin);

    gemm_wmma<0, 4><<<gemm_blocks(T, 1024, 4), 128, 0, st>>>(
        s.up, 256, p.in_proj, nullptr, s.xz, 1024, T, 1024, 256);

    dwconv_silu_k<<<(T * D_INNER + 255) / 256, 256, 0, st>>>(
        s.xz, p.conv_w, p.conv_b, s.xhs, T);

    gemm_wmma<0, 3><<<gemm_blocks(T, 144, 3), 128, 0, st>>>(
        s.xhs, 512, p.x_proj, nullptr, s.dbc, 144, T, 144, 512);

    gemm_wmma<2, 4><<<gemm_blocks(T, 512, 4), 128, 0, st>>>(
        s.dbc, 144, p.dt_w, p.dt_b, s.dt, 512, T, 512, 16);

    int waves = NC * D_INNER;
    scan_part1<<<(waves * 32 + 255) / 256, 256, 0, st>>>(
        s.dt, s.dbc, s.xhs, p.A_log, s.Q, s.Sdt, NC);
    scan_part2<<<(D_INNER * D_STATE + 255) / 256, 256, 0, st>>>(
        p.A_log, s.Q, s.Sdt, s.Hinit, NC);
    scan_part3<<<(waves * 32 + 255) / 256, 256, 0, st>>>(
        s.dt, s.dbc, s.xhs, p.A_log, s.Hinit, s.ys, NC);

    combine_k<<<(T * D_INNER + 255) / 256, 256, 0, st>>>(
        s.ys, s.xhs, p.D, s.xz, T);

    gemm_wmma<0, 4><<<gemm_blocks(T, 256, 4), 128, 0, st>>>(
        s.ys, 512, p.out_proj, nullptr, s.out, 256, T, 256, 512);

    postconv_k<<<(T * 18 + 255) / 256, 256, 0, st>>>(
        s.out, p.post_w, p.post_b, s.h, T);

    frames_k<<<(T * 16 + 255) / 256, 256, 0, st>>>(s.h, s.fr, T);
    ola_k<<<((T - 1) * 4 + 255) / 256, 256, 0, st>>>(s.fr, audio_out, T);
}

extern "C" void kernel_launch(void* const* d_in, const int* in_sizes, int n_in,
                              void* d_out, int out_size, void* d_ws, size_t ws_size,
                              hipStream_t stream)
{
    (void)in_sizes; (void)n_in; (void)out_size; (void)ws_size;

    const float* x   = (const float*)d_in[0];
    const float* cpw = (const float*)d_in[1];
    const float* cpb = (const float*)d_in[2];

    StageParams sp[2];
    for (int s = 0; s < 2; ++s) {
        int o = 3 + s * 13;
        sp[s].up_w     = (const float*)d_in[o + 0];
        sp[s].up_b     = (const float*)d_in[o + 1];
        sp[s].in_proj  = (const float*)d_in[o + 2];
        sp[s].conv_w   = (const float*)d_in[o + 3];
        sp[s].conv_b   = (const float*)d_in[o + 4];
        sp[s].x_proj   = (const float*)d_in[o + 5];
        sp[s].dt_w     = (const float*)d_in[o + 6];
        sp[s].dt_b     = (const float*)d_in[o + 7];
        sp[s].A_log    = (const float*)d_in[o + 8];
        sp[s].D        = (const float*)d_in[o + 9];
        sp[s].out_proj = (const float*)d_in[o + 10];
        sp[s].post_w   = (const float*)d_in[o + 11];
        sp[s].post_b   = (const float*)d_in[o + 12];
    }

    float* w = (float*)d_ws;
    Scratch sc;
    size_t off = 0;
    sc.up    = w + off; off += (size_t)8192 * 256;
    sc.xz    = w + off; off += (size_t)8192 * 1024;
    sc.xhs   = w + off; off += (size_t)8192 * 512;
    sc.dbc   = w + off; off += (size_t)8192 * 144;
    sc.dt    = w + off; off += (size_t)8192 * 512;
    sc.ys    = w + off; off += (size_t)8192 * 512;
    sc.out   = w + off; off += (size_t)8192 * 256;
    sc.h     = w + off; off += (size_t)8192 * 18;
    sc.fr    = w + off; off += (size_t)8192 * 16;
    sc.y0    = w + off; off += (size_t)128 * 256;
    sc.Q     = w + off; off += (size_t)64 * D_INNER * D_STATE;
    sc.Hinit = w + off; off += (size_t)64 * D_INNER * D_STATE;
    sc.Sdt   = w + off; off += (size_t)64 * D_INNER;

    float* out = (float*)d_out;

    conv_pre_k<<<128, 256, 0, stream>>>(x, cpw, cpb, sc.y0);

    // Stage 0: T 128 -> 1024, audio 4092 samples.
    run_stage(sc.y0, 128, sp[0], sc, out, stream);

    // Stage 1: input is stage-0 mamba output (sc.out, T=1024); consumed by the
    // stage-1 upsample before sc.out is overwritten (stream-ordered).
    run_stage(sc.out, 1024, sp[1], sc, out + 4092, stream);
}
